// MultiHeadAttention_39444979646815
// MI455X (gfx1250) — matrix operand order repaired and checked by reading, not compiled, not run on hardware
//
#include <hip/hip_runtime.h>
#include <hip/hip_bf16.h>

typedef __attribute__((ext_vector_type(16))) __bf16 v16bf;
typedef __attribute__((ext_vector_type(8)))  __bf16 v8bf;
typedef __attribute__((ext_vector_type(8)))  float  v8f;

constexpr int   kE     = 384;
constexpr int   kHeads = 8;
constexpr int   kB     = 8;
constexpr int   kHW    = 32;
constexpr int   kHP    = 34;            // spatial dim padded with 1-px zero halo
constexpr int   kN     = 1025;          // HW*HW + 1
constexpr int   kHD    = 48;            // E / HEADS
constexpr int   kNP    = 1056;          // kN padded to multiple of 32 (66*16)
constexpr int   kK9    = 9 * kE;        // 3456 (conv im2col K)
constexpr float kScale = 0.14433756729740643f;  // 48^-0.5

// ---------- WMMA helpers (CDNA5 wave32 layouts) ----------

__device__ inline v8f wmma_bf16(v16bf a, v16bf b, v8f c) {
  // D = A(16x32 bf16) * B(32x16 bf16) + C(16x16 f32)
  return __builtin_amdgcn_wmma_f32_16x16x32_bf16(
      /*neg_a=*/false, a, /*neg_b=*/false, b,
      /*c_mod=*/(short)0, c, /*reuse_a=*/false, /*reuse_b=*/false);
}

// A-matrix 16x32 bf16 fragment: lane holds row M=lane%16.
// elements 0..7  -> K = 2*koff + 0..7    (contiguous)
// elements 8..15 -> K = 2*koff + 8..15   (contiguous)
// koff = 8*(lane>=16). `base` points at row start + k-chunk base.
__device__ inline v16bf load_a_frag(const __bf16* base, int koff) {
  v8bf lo = *(const v8bf*)(base + 2 * koff);
  v8bf hi = *(const v8bf*)(base + 2 * koff + 8);
  v16bf r;
#pragma unroll
  for (int i = 0; i < 8; ++i) { r[i] = lo[i]; r[i + 8] = hi[i]; }
  return r;
}

__device__ inline v8f fzero8() {
  v8f z;
#pragma unroll
  for (int i = 0; i < 8; ++i) z[i] = 0.0f;
  return z;
}

// ---------- 0: zero padded buffers ----------
__global__ void k_zero(uint4* p, long n16) {
  long i = (long)blockIdx.x * blockDim.x + threadIdx.x;
  long st = (long)gridDim.x * blockDim.x;
  uint4 z; z.x = z.y = z.z = z.w = 0u;
  for (; i < n16; i += st) p[i] = z;
}

// ---------- 1: spectral-norm scale factors ----------
// scale = s_init / || Wm @ normalize(Wm^T u) ||
__global__ void k_sigma(const float* wqk, const float* wv, const float* wout,
                        const float* uqk, const float* uv, const float* uout,
                        const float* sqk, const float* sv, const float* sout,
                        float* scales) {
  __shared__ float v1[kK9];     // up to 3456 cols
  __shared__ float red[256];
  const float* wm; const float* u; const float* s0; int cols;
  if (blockIdx.x == 0)      { wm = wqk;  u = uqk;  s0 = sqk;  cols = kK9; }
  else if (blockIdx.x == 1) { wm = wv;   u = uv;   s0 = sv;   cols = kK9; }
  else                      { wm = wout; u = uout; s0 = sout; cols = kE;  }
  const int tid = threadIdx.x;

  float ss = 0.0f;
  for (int c = tid; c < cols; c += 256) {
    float t = 0.0f;
    for (int r = 0; r < kE; ++r) t += wm[(long)r * cols + c] * u[r];
    v1[c] = t; ss += t * t;
  }
  red[tid] = ss; __syncthreads();
  for (int w = 128; w > 0; w >>= 1) { if (tid < w) red[tid] += red[tid + w]; __syncthreads(); }
  float inv = 1.0f / fmaxf(sqrtf(red[0]), 1e-12f);
  __syncthreads();

  float ss2 = 0.0f;
  for (int r = tid; r < kE; r += 256) {
    float t = 0.0f;
    for (int c = 0; c < cols; ++c) t += wm[(long)r * cols + c] * v1[c];
    t *= inv; ss2 += t * t;
  }
  red[tid] = ss2; __syncthreads();
  for (int w = 128; w > 0; w >>= 1) { if (tid < w) red[tid] += red[tid + w]; __syncthreads(); }
  if (tid == 0) scales[blockIdx.x] = s0[0] / fmaxf(sqrtf(red[0]), 1e-12f);
}

// ---------- 2: bf16 prep (scaled weights, halo-padded NHWC x, CLS rows) ----------
__global__ void k_prep(const float* __restrict__ x,
                       const float* __restrict__ wqk, const float* __restrict__ wv,
                       const float* __restrict__ wout, const float* __restrict__ scales,
                       __bf16* __restrict__ wqbf, __bf16* __restrict__ wvbf,
                       __bf16* __restrict__ wobf, __bf16* __restrict__ xbf,
                       __bf16* __restrict__ Qbf, __bf16* __restrict__ Vt) {
  const long tid0 = (long)blockIdx.x * blockDim.x + threadIdx.x;
  const long gs = (long)gridDim.x * blockDim.x;
  const float s_qk = scales[0], s_v = scales[1], s_o = scales[2];

  // conv weights: w[eo][ei][kh][kw] -> wbf[kh*3+kw][eo][ei], scaled
  for (long i = tid0; i < (long)9 * kE * kE; i += gs) {
    int ei = (int)(i % kE); long t = i / kE;
    int eo = (int)(t % kE); int kk = (int)(t / kE);
    long src = ((long)eo * kE + ei) * 9 + kk;
    wqbf[i] = (__bf16)(wqk[src] * s_qk);
    wvbf[i] = (__bf16)(wv[src]  * s_v);
  }
  // output projection weight (row e, col e'), scaled
  for (long i = tid0; i < (long)kE * kE; i += gs)
    wobf[i] = (__bf16)(wout[i] * s_o);
  // x (B,N,E) spatial tokens -> halo-padded NHWC bf16 [b][1+h][1+w][e]
  // (halo border stays zero from k_zero, so conv loads need no masking)
  for (long i = tid0; i < (long)kB * kHW * kHW * kE; i += gs) {
    int e = (int)(i % kE); long t = i / kE;
    int pix = (int)(t % (kHW * kHW)); int b = (int)(t / (kHW * kHW));
    int ph = pix >> 5, pw = pix & 31;
    xbf[((long)(b * kHP + 1 + ph) * kHP + 1 + pw) * kE + e] =
        (__bf16)(x[((long)b * kN + 1 + pix) * kE + e]);
  }
  // CLS token -> Q row 0 / Vt col 0
  for (long i = tid0; i < (long)kB * kE; i += gs) {
    int e = (int)(i % kE); int b = (int)(i / kE);
    int h = e / kHD, d = e % kHD;
    float v = x[(long)b * kN * kE + e];
    int bh = b * kHeads + h;
    Qbf[((long)bh * kNP + 0) * 64 + d] = (__bf16)v;
    Vt[((long)bh * 64 + d) * kNP + 0]  = (__bf16)v;
  }
}

// ---------- 3: conv3x3 as 9 shifted WMMA GEMMs over halo-padded input ----------
// grid = (512 pixel-tiles, 3 m-groups, 2 convs), block = 256 (8 waves)
__global__ void k_conv(const __bf16* __restrict__ xbf,
                       const __bf16* __restrict__ wqbf, const __bf16* __restrict__ wvbf,
                       __bf16* __restrict__ Qbf, __bf16* __restrict__ Vt) {
  const int lane = threadIdx.x & 31;
  const int wid  = threadIdx.x >> 5;
  const int mtile = blockIdx.y * 8 + wid;          // 0..23 (out-channel tile)
  const int nt    = blockIdx.x;                    // 0..511 (pixel tile)
  const int isv   = blockIdx.z;
  const __bf16* wbf = isv ? wvbf : wqbf;
  const int b   = nt >> 6;
  const int pix = ((nt & 63) << 4) + (lane & 15);  // this lane's output pixel
  const int ph = pix >> 5, pw = pix & 31;
  const int hi = lane >> 4;
  const int koff = hi ? 8 : 0;
  const int arow = mtile * 16 + (lane & 15);       // e_out row for A loads

  v8f acc = fzero8();
  for (int kh = 0; kh < 3; ++kh) {
    for (int kw = 0; kw < 3; ++kw) {
      // padded source pixel: (ph+kh-1)+1 = ph+kh, (pw+kw-1)+1 = pw+kw — always in-bounds
      const __bf16* xrow = xbf + ((long)(b * kHP + ph + kh) * kHP + pw + kw) * kE + 16 * hi;
      const __bf16* wrow = wbf + ((long)(kh * 3 + kw) * kE + arow) * kE;
#pragma unroll
      for (int kb = 0; kb < kE; kb += 32) {
        v16bf a  = load_a_frag(wrow + kb, koff);
        v16bf bb = *(const v16bf*)(xrow + kb);
        acc = wmma_bf16(a, bb, acc);
      }
    }
  }
  // scatter D: row M = r + 8*hi (e_out), col N = lane%16 (pixel)
#pragma unroll
  for (int r = 0; r < 8; ++r) {
    int eo = mtile * 16 + r + 8 * hi;
    int h = eo / kHD, d = eo % kHD;
    __bf16 val = (__bf16)acc[r];
    int bh = b * kHeads + h;
    if (isv) Vt[((long)bh * 64 + d) * kNP + 1 + pix] = val;
    else     Qbf[((long)bh * kNP + 1 + pix) * 64 + d] = val;
  }
}

// ---------- 4: ||q||^2 per (b,h,row) ----------
__global__ void k_sq(const __bf16* __restrict__ Qbf, float* __restrict__ sq) {
  long i = (long)blockIdx.x * blockDim.x + threadIdx.x;
  if (i >= (long)kB * kHeads * kNP) return;
  const __bf16* p = Qbf + i * 64;
  float s = 0.0f;
#pragma unroll
  for (int d = 0; d < 64; ++d) { float v = (float)p[d]; s += v * v; }
  sq[i] = s;
}

// ---------- 5: distance attention, flash-style (online softmax + WMMA PV) ----------
// grid = 528 blocks * 8 waves = 64 (b,h) * 66 row-tiles
__global__ void k_attn(const __bf16* __restrict__ Qbf, const __bf16* __restrict__ Vt,
                       const float* __restrict__ sq, __bf16* __restrict__ AObf) {
  __shared__ alignas(64) __bf16 pbuf[8][16][32];   // per-wave prob tile (C->A relayout)
  const int lane = threadIdx.x & 31;
  const int wid  = threadIdx.x >> 5;
  const int gw   = blockIdx.x * 8 + wid;           // 0..4223
  const int rt   = gw % (kNP / 16);                // row tile 0..65
  const int bh   = gw / (kNP / 16);                // 0..63
  const int b = bh / kHeads, h = bh % kHeads;
  const int hi = lane >> 4;
  const int koff = hi ? 8 : 0;
  const int l16 = lane & 15;

  const __bf16* Qb  = Qbf + (long)bh * kNP * 64;
  const float*  sqb = sq  + (long)bh * kNP;

  // A fragments for this wave's 16 query rows (K = 64 = 2 chunks)
  const __bf16* qrow = Qb + (long)(rt * 16 + l16) * 64;
  v16bf aq0 = load_a_frag(qrow + 0,  koff);
  v16bf aq1 = load_a_frag(qrow + 32, koff);

  float sqr[8];
#pragma unroll
  for (int r = 0; r < 8; ++r) sqr[r] = sqb[rt * 16 + r + 8 * hi];

  v8f o0 = fzero8(), o1 = fzero8(), o2 = fzero8();
  float rmax[8], rsum[8];
#pragma unroll
  for (int r = 0; r < 8; ++r) { rmax[r] = -1e30f; rsum[r] = 0.0f; }

  for (int mb = 0; mb < kNP; mb += 32) {
    float lg[2][8];
#pragma unroll
    for (int sub = 0; sub < 2; ++sub) {
      const int colrow = mb + sub * 16 + l16;      // key index for this lane's column
      const float sqc = sqb[colrow];
      const __bf16* crow = Qb + (long)colrow * 64;
      v16bf b0 = *(const v16bf*)(crow + 16 * hi);
      v16bf b1 = *(const v16bf*)(crow + 32 + 16 * hi);
      v8f s = fzero8();
      s = wmma_bf16(aq0, b0, s);
      s = wmma_bf16(aq1, b1, s);
      const bool mvalid = colrow < kN;
#pragma unroll
      for (int r = 0; r < 8; ++r) {
        float d2 = sqr[r] + sqc - 2.0f * s[r];
        float dist = d2 > 0.0f ? sqrtf(d2) : 0.0f;
        lg[sub][r] = mvalid ? (-dist * kScale) : -1e30f;
      }
    }
    // online softmax per row (row spans 16 lanes of a half-wave)
#pragma unroll
    for (int r = 0; r < 8; ++r) {
      float tm = fmaxf(lg[0][r], lg[1][r]);
      tm = fmaxf(tm, __shfl_xor(tm, 1, 32));
      tm = fmaxf(tm, __shfl_xor(tm, 2, 32));
      tm = fmaxf(tm, __shfl_xor(tm, 4, 32));
      tm = fmaxf(tm, __shfl_xor(tm, 8, 32));
      float nm = fmaxf(rmax[r], tm);
      float corr = __expf(rmax[r] - nm);
      rmax[r] = nm;
      float p0 = __expf(lg[0][r] - nm);
      float p1 = __expf(lg[1][r] - nm);
      float ps = p0 + p1;
      ps += __shfl_xor(ps, 1, 32);
      ps += __shfl_xor(ps, 2, 32);
      ps += __shfl_xor(ps, 4, 32);
      ps += __shfl_xor(ps, 8, 32);
      rsum[r] = rsum[r] * corr + ps;
      o0[r] *= corr; o1[r] *= corr; o2[r] *= corr;
      const int rowM = r + 8 * hi;                 // C-layout row
      pbuf[wid][rowM][l16]      = (__bf16)p0;
      pbuf[wid][rowM][16 + l16] = (__bf16)p1;
    }
    // reload probs in A layout (per-wave LDS, DS ops are in-order within a wave)
    v16bf ap = load_a_frag(&pbuf[wid][l16][0], koff);
    // V^T b-frags: lane column n = d-tile*16 + l16, k walks m (contiguous)
    const __bf16* vb = Vt + ((long)bh * 64 + l16) * kNP + mb + 16 * hi;
    v16bf bv0 = *(const v16bf*)(vb);
    v16bf bv1 = *(const v16bf*)(vb + (long)16 * kNP);
    v16bf bv2 = *(const v16bf*)(vb + (long)32 * kNP);
    o0 = wmma_bf16(ap, bv0, o0);
    o1 = wmma_bf16(ap, bv1, o1);
    o2 = wmma_bf16(ap, bv2, o2);
  }

#pragma unroll
  for (int r = 0; r < 8; ++r) {
    const int n = rt * 16 + r + 8 * hi;
    if (n < kN) {
      const float inv = 1.0f / rsum[r];
      const long base = ((long)b * kNP + n) * kE + h * kHD;
      AObf[base + l16]      = (__bf16)(o0[r] * inv);
      AObf[base + 16 + l16] = (__bf16)(o1[r] * inv);
      AObf[base + 32 + l16] = (__bf16)(o2[r] * inv);
    }
  }
}

// ---------- 6: output projection out = AO @ Wo^T + b ----------
// grid = (66, 24), block = 256 (8 waves: 8 m-subtiles)
__global__ void k_proj(const __bf16* __restrict__ AObf, const __bf16* __restrict__ wobf,
                       const float* __restrict__ bout, float* __restrict__ out) {
  const int lane = threadIdx.x & 31;
  const int wid  = threadIdx.x >> 5;
  const int mt  = blockIdx.x * 8 + wid;            // 0..527 (row tile over B*NP)
  const int ntl = blockIdx.y;                      // 0..23  (col tile over E)
  const int hi = lane >> 4;
  const int koff = hi ? 8 : 0;
  const int l16 = lane & 15;

  const __bf16* arow = AObf + (long)(mt * 16 + l16) * kE;
  const __bf16* brow = wobf + (long)(ntl * 16 + l16) * kE;   // B[k=e'][n=e] = wo[e][e']
  v8f acc = fzero8();
#pragma unroll
  for (int kb = 0; kb < kE; kb += 32) {
    v16bf a  = load_a_frag(arow + kb, koff);
    v16bf bb = *(const v16bf*)(brow + kb + 16 * hi);
    acc = wmma_bf16(a, bb, acc);
  }
  const int e = ntl * 16 + l16;
  const float bo = bout[e];
#pragma unroll
  for (int r = 0; r < 8; ++r) {
    const int row = mt * 16 + r + 8 * hi;
    const int b = row / kNP, n = row % kNP;
    if (n < kN) out[((long)b * kN + n) * kE + e] = acc[r] + bo;
  }
}

// ---------- launcher ----------
extern "C" void kernel_launch(void* const* d_in, const int* in_sizes, int n_in,
                              void* d_out, int out_size, void* d_ws, size_t ws_size,
                              hipStream_t stream) {
  const float* x    = (const float*)d_in[0];
  const float* wqk  = (const float*)d_in[1];
  const float* wv   = (const float*)d_in[2];
  const float* wout = (const float*)d_in[3];
  const float* bout = (const float*)d_in[4];
  const float* uqk  = (const float*)d_in[5];
  const float* sqk  = (const float*)d_in[7];
  const float* uv   = (const float*)d_in[8];
  const float* sv   = (const float*)d_in[10];
  const float* uout = (const float*)d_in[11];
  const float* sout = (const float*)d_in[13];
  float* out = (float*)d_out;

  char* ws = (char*)d_ws;
  size_t off = 0;
  auto take = [&](size_t bytes) -> char* {
    char* p = ws + off;
    off = (off + bytes + 255) & ~(size_t)255;
    return p;
  };
  float*  scales = (float*)take(16);
  __bf16* wqbf = (__bf16*)take((size_t)9 * kE * kE * 2);
  __bf16* wvbf = (__bf16*)take((size_t)9 * kE * kE * 2);
  __bf16* wobf = (__bf16*)take((size_t)kE * kE * 2);
  float*  sqb  = (float*) take((size_t)kB * kHeads * kNP * 4);
  // zero-initialized span (contiguous): xbf (halo), Qbf, Vt, AObf
  char* zbase = ws + off;
  const size_t xbytes = (size_t)kB * kHP * kHP * kE * 2;
  __bf16* xbf  = (__bf16*)take(xbytes);
  __bf16* Qbf  = (__bf16*)take((size_t)kB * kHeads * kNP * 64 * 2);
  __bf16* Vt   = (__bf16*)take((size_t)kB * kHeads * 64 * kNP * 2);
  __bf16* AObf = (__bf16*)take((size_t)kB * kNP * kE * 2);
  const size_t zbytes = xbytes
                      + (size_t)kB * kHeads * kNP * 64 * 2 * 2
                      + (size_t)kB * kNP * kE * 2;

  k_zero<<<2048, 256, 0, stream>>>((uint4*)zbase, (long)(zbytes / 16));
  k_sigma<<<3, 256, 0, stream>>>(wqk, wv, wout, uqk, uv, uout, sqk, sv, sout, scales);
  k_prep<<<1024, 256, 0, stream>>>(x, wqk, wv, wout, scales, wqbf, wvbf, wobf, xbf, Qbf, Vt);
  k_conv<<<dim3(512, 3, 2), 256, 0, stream>>>(xbf, wqbf, wvbf, Qbf, Vt);
  k_sq<<<(kB * kHeads * kNP + 255) / 256, 256, 0, stream>>>(Qbf, sqb);
  k_attn<<<528, 256, 0, stream>>>(Qbf, Vt, sqb, AObf);
  k_proj<<<dim3(66, 24), 256, 0, stream>>>(AObf, wobf, bout, out);
}
